// GraphAttentionLayer_11776800326119
// MI455X (gfx1250) — matrix operand order repaired and checked by reading, not compiled, not run on hardware
//
#include <hip/hip_runtime.h>

// ---------------------------------------------------------------------------
// GAT layer for MI455X (gfx1250), wave32, bf16 WMMA, flash-style attention.
// B=8, N=2048, F_in=512, F_out=256  (sizes hardcoded from the reference)
// ---------------------------------------------------------------------------

typedef __attribute__((ext_vector_type(16))) __bf16        v16bf;
typedef __attribute__((ext_vector_type(8)))  float         v8f;
typedef __attribute__((ext_vector_type(8)))  unsigned int  v8u;

#define BATCH   8
#define NNODE   2048
#define FIN     512
#define FOUT    256
#define MROWS   (BATCH * NNODE)      // 16384
#define ALPHA   0.2f
#define NEGINF  (-9.0e15f)

__device__ __forceinline__ unsigned short f2bf(float f) {
    unsigned int u = __float_as_uint(f);
    unsigned int r = (u + 0x7FFFu + ((u >> 16) & 1u)) >> 16;   // RNE
    return (unsigned short)r;
}
__device__ __forceinline__ float bf2f(unsigned short s) {
    return __uint_as_float(((unsigned int)s) << 16);
}

// 8 paired-bf16 dword loads; offsets are contiguous per the WMMA operand
// swizzle, so these merge into two global_load_b128 per call.
__device__ __forceinline__ v8u load8g(const unsigned short* p, const int* off) {
    v8u u;
#pragma unroll
    for (int v = 0; v < 8; ++v)
        u[v] = *(const unsigned int*)(p + off[v]);
    return u;
}

// ---------------------------------------------------------------------------
// Kernel 1a: h (f32) -> h bf16, flat copy
// ---------------------------------------------------------------------------
__global__ __launch_bounds__(256) void cvt_h_kernel(const float* __restrict__ h,
                                                    unsigned short* __restrict__ hB,
                                                    int n) {
    int i = blockIdx.x * blockDim.x + threadIdx.x;
    if (i < n) hB[i] = f2bf(h[i]);
}

// ---------------------------------------------------------------------------
// Kernel 1b: W (512x256 f32) -> Wt (256x512 bf16, f-major)
// ---------------------------------------------------------------------------
__global__ __launch_bounds__(256) void cvt_W_kernel(const float* __restrict__ W,
                                                    unsigned short* __restrict__ Wt) {
    int i = blockIdx.x * blockDim.x + threadIdx.x;     // coalesced read
    int k = i >> 8;           // 0..511
    int f = i & 255;          // 0..255
    Wt[(size_t)f * FIN + k] = f2bf(W[i]);
}

// ---------------------------------------------------------------------------
// Kernel 2: Wh = h @ W with v_wmma_f32_16x16x32_bf16.
// Block = 128 threads (4 waves). Wave w: rows [16*blockIdx.x,+16),
// cols [64*w,+64), K-loop 512/32 = 16 steps.
// Software-pipelined one tile ahead so WMMA overlaps VMEM (partial waits).
// ---------------------------------------------------------------------------
__global__ __launch_bounds__(128) void gemm1_kernel(const unsigned short* __restrict__ hB,
                                                    const unsigned short* __restrict__ Wt,
                                                    unsigned short* __restrict__ Whr,
                                                    unsigned short* __restrict__ Wht) {
    const int lane = threadIdx.x & 31;
    const int wave = threadIdx.x >> 5;      // 0..3 -> 64-col quad
    const int m0   = blockIdx.x * 16;       // row tile base (flat rows)
    const int Ml   = lane & 15;
    const int hi   = lane >> 4;
    const int colq = wave * 64;

    // A layout (16-bit 16x32): v<4 -> K=2v,2v+1 ; v>=4 -> K=16+2(v-4),+1 ; +8 hi
    int aK[8];
#pragma unroll
    for (int v = 0; v < 8; ++v)
        aK[v] = (v < 4 ? 2 * v : 16 + 2 * (v - 4)) + hi * 8;
    // B layout (16-bit 32x16): same K order as A (v<4 -> K=2v,2v+1 ; v>=4 -> K=16+2(v-4),+1 ; +8 hi)
    int bK[8];
#pragma unroll
    for (int v = 0; v < 8; ++v)
        bK[v] = (v < 4 ? 2 * v : 16 + 2 * (v - 4)) + hi * 8;

    const unsigned short* arow = hB + (size_t)(m0 + Ml) * FIN;
    const unsigned short* bbase[4];
#pragma unroll
    for (int t = 0; t < 4; ++t)
        bbase[t] = Wt + (size_t)(colq + t * 16 + Ml) * FIN;

    v8f acc[4];
#pragma unroll
    for (int t = 0; t < 4; ++t)
#pragma unroll
        for (int r = 0; r < 8; ++r) acc[t][r] = 0.0f;

    // prologue of the pipeline
    v8u au = load8g(arow, aK);
    v8u bu = load8g(bbase[0], bK);

    for (int s = 0; s < FIN / 32; ++s) {
        const int k0 = s * 32;
        const int kn = (s == FIN / 32 - 1) ? 0 : k0 + 32;   // wrap: harmless reload
        v16bf A = __builtin_bit_cast(v16bf, au);

        v8u b1 = load8g(bbase[1] + k0, bK);
        acc[0] = __builtin_amdgcn_wmma_f32_16x16x32_bf16(
            false, A, false, __builtin_bit_cast(v16bf, bu), (short)0, acc[0], false, false);

        v8u b2 = load8g(bbase[2] + k0, bK);
        acc[1] = __builtin_amdgcn_wmma_f32_16x16x32_bf16(
            false, A, false, __builtin_bit_cast(v16bf, b1), (short)0, acc[1], false, false);

        v8u b3 = load8g(bbase[3] + k0, bK);
        v8u an = load8g(arow + kn, aK);
        acc[2] = __builtin_amdgcn_wmma_f32_16x16x32_bf16(
            false, A, false, __builtin_bit_cast(v16bf, b2), (short)0, acc[2], false, false);

        v8u b0n = load8g(bbase[0] + kn, bK);
        acc[3] = __builtin_amdgcn_wmma_f32_16x16x32_bf16(
            false, A, false, __builtin_bit_cast(v16bf, b3), (short)0, acc[3], false, false);

        au = an;
        bu = b0n;
    }

    // C layout: VGPR r -> row M = r + 8*hi, col N = lane%16
    const int b     = m0 >> 11;
    const int nbase = m0 & 2047;
#pragma unroll
    for (int t = 0; t < 4; ++t) {
        const int f = colq + t * 16 + Ml;
#pragma unroll
        for (int r = 0; r < 8; ++r) {
            const int row = r + hi * 8;
            unsigned short bf = f2bf(acc[t][r]);
            Whr[(size_t)(m0 + row) * FOUT + f] = bf;
            Wht[((size_t)b * FOUT + f) * NNODE + (nbase + row)] = bf;
        }
    }
}

// ---------------------------------------------------------------------------
// Kernel 3: f1/f2 = Wh . a1 / Wh . a2, one 256-thread block per row.
// ---------------------------------------------------------------------------
__global__ __launch_bounds__(256) void f_kernel(const unsigned short* __restrict__ Whr,
                                                const float* __restrict__ a,
                                                float* __restrict__ f1,
                                                float* __restrict__ f2) {
    __shared__ float s1[256];
    __shared__ float s2[256];
    const int row = blockIdx.x;
    const int t   = threadIdx.x;
    float v = bf2f(Whr[(size_t)row * FOUT + t]);
    s1[t] = v * a[t];
    s2[t] = v * a[FOUT + t];
    __syncthreads();
#pragma unroll
    for (int off = 128; off > 0; off >>= 1) {
        if (t < off) { s1[t] += s1[t + off]; s2[t] += s2[t + off]; }
        __syncthreads();
    }
    if (t == 0) { f1[row] = s1[0]; f2[row] = s2[0]; }
}

// ---------------------------------------------------------------------------
// Kernel 4: fused masked-softmax attention (flash style).
// Block = 128 threads (4 waves) handles 16 rows x all 256 output features.
// The four Wht B-tiles are loaded at the TOP of each j-iteration: they only
// depend on j0, and VMEM loads stay outstanding across the softmax barriers,
// so the whole score/softmax phase hides their latency.
// ---------------------------------------------------------------------------
__global__ __launch_bounds__(128) void attn_kernel(const int* __restrict__ adj,
                                                   const unsigned short* __restrict__ Wht,
                                                   const float* __restrict__ f1,
                                                   const float* __restrict__ f2,
                                                   float* __restrict__ out) {
    __shared__ float          sS[16][33];      // scores / probs (padded)
    __shared__ unsigned short sP[16][32];      // bf16 probs, [M][K]
    __shared__ float          sM[16], sL[16], sScale[16], sF1[16];

    const int tid  = threadIdx.x;
    const int lane = tid & 31;
    const int wave = tid >> 5;
    const int Ml   = lane & 15;
    const int hi   = lane >> 4;

    const int b  = blockIdx.x >> 7;            // 2048/16 = 128 tiles per batch
    const int i0 = (blockIdx.x & 127) << 4;

    if (tid < 16) {
        sM[tid]  = -3.0e38f;
        sL[tid]  = 0.0f;
        sF1[tid] = f1[(size_t)b * NNODE + i0 + tid];
    }

    v8f acc[4];
#pragma unroll
    for (int t = 0; t < 4; ++t)
#pragma unroll
        for (int r = 0; r < 8; ++r) acc[t][r] = 0.0f;

    const int r   = tid >> 3;                  // score row this thread fills
    const int jj0 = (tid & 7) << 2;            // 4 consecutive j per thread
    const int*   adjRow = adj + ((size_t)b * NNODE + i0 + r) * NNODE;
    const float* f2b    = f2 + (size_t)b * NNODE;

    int aK[8];
#pragma unroll
    for (int v = 0; v < 8; ++v)
        aK[v] = (v < 4 ? 2 * v : 16 + 2 * (v - 4)) + hi * 8;
    int bK[8];
#pragma unroll
    for (int v = 0; v < 8; ++v)
        bK[v] = (v < 4 ? 2 * v : 16 + 2 * (v - 4)) + hi * 8;

    const unsigned short* bbase[4];
#pragma unroll
    for (int t = 0; t < 4; ++t)
        bbase[t] = Wht + ((size_t)b * FOUT + (wave * 64 + t * 16 + Ml)) * NNODE;

    __syncthreads();

    for (int j0 = 0; j0 < NNODE; j0 += 32) {
        // -- phase 0: kick off all four B-tile load clauses; they complete
        //    while the softmax phases below run (loads survive barriers). --
        v8u bu0 = load8g(bbase[0] + j0, bK);
        v8u bu1 = load8g(bbase[1] + j0, bK);
        v8u bu2 = load8g(bbase[2] + j0, bK);
        v8u bu3 = load8g(bbase[3] + j0, bK);

        // -- phase 1: scores (coalesced int4/float4; adj read exactly once) --
        if (j0 + 32 < NNODE)
            __builtin_prefetch(adjRow + j0 + 32 + jj0, 0, 1);
        int4   am = *(const int4*)(adjRow + j0 + jj0);
        float4 fv = *(const float4*)(f2b + j0 + jj0);
        float  fr = sF1[r];
        float  s0 = fr + fv.x, s1 = fr + fv.y, s2 = fr + fv.z, s3 = fr + fv.w;
        s0 = s0 > 0.0f ? s0 : ALPHA * s0;  s0 = am.x > 0 ? s0 : NEGINF;
        s1 = s1 > 0.0f ? s1 : ALPHA * s1;  s1 = am.y > 0 ? s1 : NEGINF;
        s2 = s2 > 0.0f ? s2 : ALPHA * s2;  s2 = am.z > 0 ? s2 : NEGINF;
        s3 = s3 > 0.0f ? s3 : ALPHA * s3;  s3 = am.w > 0 ? s3 : NEGINF;
        sS[r][jj0 + 0] = s0; sS[r][jj0 + 1] = s1;
        sS[r][jj0 + 2] = s2; sS[r][jj0 + 3] = s3;
        __syncthreads();

        // -- phase 2a: running max + rescale factor (16 threads) --
        if (tid < 16) {
            float mo = sM[tid];
            float mt = mo;
#pragma unroll
            for (int k = 0; k < 32; ++k) mt = fmaxf(mt, sS[tid][k]);
            sScale[tid] = __expf(mo - mt);
            sM[tid]     = mt;
        }
        __syncthreads();

        // -- phase 2b: exponentials by all 128 threads --
        {
            float mt = sM[r];
            float p0 = __expf(sS[r][jj0 + 0] - mt);
            float p1 = __expf(sS[r][jj0 + 1] - mt);
            float p2 = __expf(sS[r][jj0 + 2] - mt);
            float p3 = __expf(sS[r][jj0 + 3] - mt);
            sS[r][jj0 + 0] = p0; sP[r][jj0 + 0] = f2bf(p0);
            sS[r][jj0 + 1] = p1; sP[r][jj0 + 1] = f2bf(p1);
            sS[r][jj0 + 2] = p2; sP[r][jj0 + 2] = f2bf(p2);
            sS[r][jj0 + 3] = p3; sP[r][jj0 + 3] = f2bf(p3);
        }
        __syncthreads();

        // -- phase 2c: denominator update (16 threads); no barrier needed
        //    before phase 3 (phase 3 touches neither sS nor sL). --
        if (tid < 16) {
            float l = sL[tid] * sScale[tid];
#pragma unroll
            for (int k = 0; k < 32; ++k) l += sS[tid][k];
            sL[tid] = l;
        }

        // -- phase 3: rescale accumulators, then P @ Wh via WMMA --
        float scr[8];
#pragma unroll
        for (int r8 = 0; r8 < 8; ++r8) scr[r8] = sScale[r8 + hi * 8];
#pragma unroll
        for (int t = 0; t < 4; ++t)
#pragma unroll
            for (int r8 = 0; r8 < 8; ++r8) acc[t][r8] *= scr[r8];

        v8u au;
#pragma unroll
        for (int v = 0; v < 8; ++v)
            au[v] = *(const unsigned int*)(&sP[Ml][aK[v]]);
        v16bf A = __builtin_bit_cast(v16bf, au);

        acc[0] = __builtin_amdgcn_wmma_f32_16x16x32_bf16(
            false, A, false, __builtin_bit_cast(v16bf, bu0), (short)0, acc[0], false, false);
        acc[1] = __builtin_amdgcn_wmma_f32_16x16x32_bf16(
            false, A, false, __builtin_bit_cast(v16bf, bu1), (short)0, acc[1], false, false);
        acc[2] = __builtin_amdgcn_wmma_f32_16x16x32_bf16(
            false, A, false, __builtin_bit_cast(v16bf, bu2), (short)0, acc[2], false, false);
        acc[3] = __builtin_amdgcn_wmma_f32_16x16x32_bf16(
            false, A, false, __builtin_bit_cast(v16bf, bu3), (short)0, acc[3], false, false);

        __syncthreads();
    }

    // -- epilogue: divide by softmax denominator, ELU, store f32 output --
    float linv[8];
#pragma unroll
    for (int r8 = 0; r8 < 8; ++r8) linv[r8] = 1.0f / sL[r8 + hi * 8];

#pragma unroll
    for (int t = 0; t < 4; ++t) {
        const int f = wave * 64 + t * 16 + Ml;
#pragma unroll
        for (int r8 = 0; r8 < 8; ++r8) {
            const int row = r8 + hi * 8;
            float x = acc[t][r8] * linv[r8];
            x = x > 0.0f ? x : (__expf(x) - 1.0f);
            out[((size_t)b * NNODE + i0 + row) * FOUT + f] = x;
        }
    }
}

// ---------------------------------------------------------------------------
// Launch
// ---------------------------------------------------------------------------
extern "C" void kernel_launch(void* const* d_in, const int* in_sizes, int n_in,
                              void* d_out, int out_size, void* d_ws, size_t ws_size,
                              hipStream_t stream) {
    (void)in_sizes; (void)n_in; (void)out_size; (void)ws_size;

    const float* h   = (const float*)d_in[0];   // [8,2048,512]
    const int*   adj = (const int*)d_in[1];     // [8,2048,2048]
    const float* W   = (const float*)d_in[2];   // [512,256]
    const float* a   = (const float*)d_in[3];   // [512,1]
    float*       out = (float*)d_out;           // [8,2048,256]

    char* ws = (char*)d_ws;
    unsigned short* hB  = (unsigned short*)(ws);                       // 16,777,216 B
    unsigned short* Wt  = (unsigned short*)(ws + 16777216);            //    262,144 B
    unsigned short* Whr = (unsigned short*)(ws + 17039360);            //  8,388,608 B
    unsigned short* Wht = (unsigned short*)(ws + 25427968);            //  8,388,608 B
    float*          f1  = (float*)(ws + 33816576);                     //     65,536 B
    float*          f2  = (float*)(ws + 33882112);                     //     65,536 B

    const int nH = BATCH * NNODE * FIN;                                // 8,388,608
    cvt_h_kernel<<<nH / 256, 256, 0, stream>>>(h, hB, nH);
    cvt_W_kernel<<<(FIN * FOUT) / 256, 256, 0, stream>>>(W, Wt);
    gemm1_kernel<<<MROWS / 16, 128, 0, stream>>>(hB, Wt, Whr, Wht);
    f_kernel<<<MROWS, 256, 0, stream>>>(Whr, a, f1, f2);
    attn_kernel<<<MROWS / 16, 128, 0, stream>>>(adj, Wht, f1, f2, out);
}
